// GraphTransformerLayer_41068477285036
// MI455X (gfx1250) — matrix operand order repaired and checked by reading, not compiled, not run on hardware
//
#include <hip/hip_runtime.h>

#define DIM 512
#define HEADS 8
#define DK 64

typedef __attribute__((ext_vector_type(16))) __bf16 v16bf;
typedef __attribute__((ext_vector_type(8)))  float  v8f;

// ---------- helpers ----------
__device__ __forceinline__ unsigned short f2bf(float f) {
    unsigned int u = __float_as_uint(f);
    u += 0x7fffu + ((u >> 16) & 1u);          // round-to-nearest-even
    return (unsigned short)(u >> 16);
}
__device__ __forceinline__ float b2f(unsigned int h) {
    return __uint_as_float(h << 16);
}
__device__ __forceinline__ v16bf pack16(const unsigned int* u) {
    union { unsigned int u[8]; v16bf v; } x;
#pragma unroll
    for (int i = 0; i < 8; ++i) x.u[i] = u[i];
    return x.v;
}

// A fragment: 16x32 bf16 from an LDS-staged 16xDIM row-major tile.
// ISA layout: lanes 0-15 M=0..15 K-base 0, lanes 16-31 same M K-base +8;
// VGPR 0-3 -> K 0..7 (+half*8), VGPR 4-7 -> K 16..23 (+half*8).
__device__ __forceinline__ v16bf load_fragA_lds(const unsigned short* As, int k0) {
    int lane = threadIdx.x & 31;
    int half = lane >> 4;
    int m    = lane & 15;
    const unsigned int* rowp = (const unsigned int*)(As + m * DIM);
    unsigned int u[8];
#pragma unroll
    for (int v = 0; v < 8; ++v) {
        int kk = (v < 4) ? (half * 8 + v * 2) : (16 + half * 8 + (v - 4) * 2);
        u[v] = rowp[(k0 + kk) >> 1];
    }
    return pack16(u);
}

// B fragment: 32x16 bf16 where B[k][n] = W[col0+n][k], W row-major [DIM x DIM].
// Layout matches A: lanes 0-15 K-base 0, lanes 16-31 K-base +8;
// VGPR 0-3 -> K 0..7 (+half*8), VGPR 4-7 -> K 16..23 (+half*8).
__device__ __forceinline__ v16bf load_fragB(const unsigned short* W, int col0, int k0) {
    int lane = threadIdx.x & 31;
    int half = lane >> 4;
    int n    = lane & 15;
    const unsigned int* rowp = (const unsigned int*)(W + (size_t)(col0 + n) * DIM);
    unsigned int u[8];
#pragma unroll
    for (int v = 0; v < 8; ++v) {
        int kk = (v < 4) ? (half * 8 + v * 2) : (16 + half * 8 + (v - 4) * 2);
        u[v] = rowp[(k0 + kk) >> 1];
    }
    return pack16(u);
}

// ---------- small utility kernels ----------
__global__ void cvt_f32_bf16(const float* __restrict__ src, unsigned short* __restrict__ dst, int n) {
    int i = blockIdx.x * blockDim.x + threadIdx.x;
    if (i < n) dst[i] = f2bf(src[i]);
}

__global__ void zero_i32(int* p, int n) {
    int i = blockIdx.x * blockDim.x + threadIdx.x;
    if (i < n) p[i] = 0;
}

// CSR build step 1: per-destination degree histogram
__global__ void hist_dst(const int* __restrict__ dst, int* __restrict__ cnt, int E) {
    int e = blockIdx.x * blockDim.x + threadIdx.x;
    if (e < E) atomicAdd(&cnt[dst[e]], 1);
}

// CSR build step 2: single-block exclusive prefix scan (n ~ 50k: trivial cost).
// Also zeroes cnt so it can serve as the scatter cursor.
__global__ void __launch_bounds__(256) scan_excl(int* __restrict__ cnt, int* __restrict__ rowptr, int n) {
    __shared__ int buf[256];
    __shared__ int carry;
    if (threadIdx.x == 0) carry = 0;
    __syncthreads();
    for (int base = 0; base < n; base += 256) {
        int i = base + (int)threadIdx.x;
        int x = (i < n) ? cnt[i] : 0;
        if (i < n) cnt[i] = 0;
        buf[threadIdx.x] = x;
        __syncthreads();
#pragma unroll
        for (int off = 1; off < 256; off <<= 1) {
            int t = (threadIdx.x >= (unsigned)off) ? buf[threadIdx.x - off] : 0;
            __syncthreads();
            buf[threadIdx.x] += t;
            __syncthreads();
        }
        int excl = carry + buf[threadIdx.x] - x;
        if (i < n) rowptr[i] = excl;
        int total = buf[255];
        __syncthreads();
        if (threadIdx.x == 0) carry += total;
        __syncthreads();
    }
    if (threadIdx.x == 0) rowptr[n] = carry;
}

// CSR build step 3: scatter edge ids into per-destination lists
__global__ void fill_csr(const int* __restrict__ dst, const int* __restrict__ rowptr,
                         int* __restrict__ cursor, int* __restrict__ eidx, int E) {
    int e = blockIdx.x * blockDim.x + threadIdx.x;
    if (e < E) {
        int d = dst[e];
        int pos = rowptr[d] + atomicAdd(&cursor[d], 1);
        eidx[pos] = e;
    }
}

// ---------- fused q/k/v projection (WMMA) ----------
// blockIdx.z selects q/k/v. Block = 256 threads (8 waves); the block's 16xDIM
// A tile (contiguous 16KB) is staged once into LDS; each wave computes a 16x64
// output strip (4 accumulators, B frags batched ahead of the WMMAs).
__global__ void __launch_bounds__(256)
qkv_gemm(const unsigned short* __restrict__ hsrc, const unsigned short* __restrict__ hdst,
         const unsigned short* __restrict__ wq, const unsigned short* __restrict__ wk,
         const unsigned short* __restrict__ wv,
         const float* __restrict__ bq, const float* __restrict__ bk, const float* __restrict__ bv,
         unsigned short* __restrict__ q, unsigned short* __restrict__ k, unsigned short* __restrict__ v) {
    __shared__ unsigned short As[16 * DIM];       // 16 KB
    int which = blockIdx.z;
    const unsigned short* A    = (which == 0) ? hdst : hsrc;
    const unsigned short* W    = (which == 0) ? wq : (which == 1 ? wk : wv);
    const float*          bias = (which == 0) ? bq : (which == 1 ? bk : bv);
    unsigned short*       out  = (which == 0) ? q  : (which == 1 ? k  : v);

    int row0 = blockIdx.x * 16;
    int wave = threadIdx.x >> 5;
    int col0 = wave * 64;

    // cooperative contiguous tile load: 1024 uint4, 256 threads x 4
    {
        const uint4* g = (const uint4*)(A + (size_t)row0 * DIM);
        uint4* s = (uint4*)As;
#pragma unroll
        for (int i = 0; i < 4; ++i) s[threadIdx.x + 256 * i] = g[threadIdx.x + 256 * i];
    }
    __syncthreads();

    v8f acc[4];
#pragma unroll
    for (int t = 0; t < 4; ++t) acc[t] = 0.0f;

    for (int k0 = 0; k0 < DIM; k0 += 32) {
        v16bf af = load_fragA_lds(As, k0);
        v16bf bfr[4];
#pragma unroll
        for (int t = 0; t < 4; ++t) bfr[t] = load_fragB(W, col0 + t * 16, k0);
#pragma unroll
        for (int t = 0; t < 4; ++t)
            acc[t] = __builtin_amdgcn_wmma_f32_16x16x32_bf16(
                false, af, false, bfr[t], (short)0, acc[t], false, false);
    }

    int lane = threadIdx.x & 31;
    int half = lane >> 4;
    int n    = lane & 15;
#pragma unroll
    for (int t = 0; t < 4; ++t) {
        int col = col0 + t * 16 + n;
        float bb = bias[col];
#pragma unroll
        for (int r = 0; r < 8; ++r) {
            int m = r + 8 * half;
            out[(size_t)(row0 + m) * DIM + col] = f2bf(acc[t][r] + bb);
        }
    }
}

// ---------- edge scores ----------
// One wave per edge: 8-head dot products k[src]·q[dst]; 4-lane shuffle reduce.
__global__ void __launch_bounds__(256)
edge_score(const unsigned short* __restrict__ kbf, const unsigned short* __restrict__ qbf,
           const int* __restrict__ src, const int* __restrict__ dst,
           float* __restrict__ score, int E) {
    int wave = threadIdx.x >> 5;
    int e = blockIdx.x * 8 + wave;
    if (e >= E) return;
    int lane = threadIdx.x & 31;
    int s = src[e], d = dst[e];
    const unsigned int* kp = (const unsigned int*)(kbf + (size_t)s * DIM) + lane * 8;
    const unsigned int* qp = (const unsigned int*)(qbf + (size_t)d * DIM) + lane * 8;
    float acc = 0.0f;
#pragma unroll
    for (int i = 0; i < 8; ++i) {
        unsigned int ku = kp[i], qu = qp[i];
        acc += b2f(ku & 0xffffu) * b2f(qu & 0xffffu) + b2f(ku >> 16) * b2f(qu >> 16);
    }
    acc += __shfl_xor(acc, 1, 32);
    acc += __shfl_xor(acc, 2, 32);
    if ((lane & 3) == 0)
        score[(size_t)e * HEADS + (lane >> 2)] = acc * 0.125f;   // / sqrt(64)
}

// ---------- per-destination softmax + weighted aggregation (no atomics) ----------
// One wave per destination node. Lanes 0-7 compute per-head max and exp-sum over
// the node's CSR edge list (exp written back in place); then all 32 lanes
// accumulate v[src]*attn into registers (16 dims/lane) and emit the bf16 row.
__global__ void __launch_bounds__(256)
dst_agg(const unsigned short* __restrict__ vbf, const int* __restrict__ src,
        const int* __restrict__ eidx, const int* __restrict__ rowptr,
        float* __restrict__ score, unsigned short* __restrict__ aggbf, int N) {
    int wave = threadIdx.x >> 5;
    int d = blockIdx.x * 8 + wave;
    if (d >= N) return;
    int lane = threadIdx.x & 31;
    int beg = rowptr[d], end = rowptr[d + 1];

    float m = __uint_as_float(0xff800000u);   // -inf
    float ssum = 0.0f;
    if (lane < 8) {
        for (int i = beg; i < end; ++i)
            m = fmaxf(m, score[(size_t)eidx[i] * HEADS + lane]);
        for (int i = beg; i < end; ++i) {
            size_t idx = (size_t)eidx[i] * HEADS + lane;
            float ex = __expf(score[idx] - m);
            ssum += ex;
            score[idx] = ex;                   // in-place, owned by this wave
        }
    }
    int head = lane >> 2;
    float sh  = __shfl(ssum, head, 32);
    float inv = (end > beg) ? 1.0f / sh : 0.0f;

    float acc[16];
#pragma unroll
    for (int t = 0; t < 16; ++t) acc[t] = 0.0f;

    for (int i = beg; i < end; ++i) {
        int e = eidx[i];
        float w = score[(size_t)e * HEADS + head] * inv;
        const unsigned int* vp = (const unsigned int*)(vbf + (size_t)src[e] * DIM) + lane * 8;
#pragma unroll
        for (int j = 0; j < 8; ++j) {
            unsigned int vu = vp[j];
            acc[2 * j]     += b2f(vu & 0xffffu) * w;
            acc[2 * j + 1] += b2f(vu >> 16) * w;
        }
    }

    unsigned int* op = (unsigned int*)(aggbf + (size_t)d * DIM) + lane * 8;
#pragma unroll
    for (int j = 0; j < 8; ++j)
        op[j] = (unsigned int)f2bf(acc[2 * j]) | ((unsigned int)f2bf(acc[2 * j + 1]) << 16);
}

// ---------- output projection + residual + LayerNorm (WMMA, fused) ----------
__global__ void __launch_bounds__(256)
proj_ln(const unsigned short* __restrict__ aggbf, const unsigned short* __restrict__ wp,
        const float* __restrict__ bp, const float* __restrict__ hdst,
        const float* __restrict__ gamma, const float* __restrict__ beta,
        float* __restrict__ out) {
    __shared__ unsigned short As[16 * DIM];   // 16 KB
    __shared__ float tile[16][DIM + 4];       // 33 KB
    int row0 = blockIdx.x * 16;
    int wave = threadIdx.x >> 5;
    int col0 = wave * 64;

    {
        const uint4* g = (const uint4*)(aggbf + (size_t)row0 * DIM);
        uint4* s = (uint4*)As;
#pragma unroll
        for (int i = 0; i < 4; ++i) s[threadIdx.x + 256 * i] = g[threadIdx.x + 256 * i];
    }
    __syncthreads();

    v8f acc[4];
#pragma unroll
    for (int t = 0; t < 4; ++t) acc[t] = 0.0f;

    for (int k0 = 0; k0 < DIM; k0 += 32) {
        v16bf af = load_fragA_lds(As, k0);
        v16bf bfr[4];
#pragma unroll
        for (int t = 0; t < 4; ++t) bfr[t] = load_fragB(wp, col0 + t * 16, k0);
#pragma unroll
        for (int t = 0; t < 4; ++t)
            acc[t] = __builtin_amdgcn_wmma_f32_16x16x32_bf16(
                false, af, false, bfr[t], (short)0, acc[t], false, false);
    }

    int lane = threadIdx.x & 31;
    int half = lane >> 4;
    int n    = lane & 15;
#pragma unroll
    for (int t = 0; t < 4; ++t) {
        int col = col0 + t * 16 + n;
        float bb = bp[col];
#pragma unroll
        for (int r = 0; r < 8; ++r) {
            int m = r + 8 * half;
            tile[m][col] = acc[t][r] + bb + hdst[(size_t)(row0 + m) * DIM + col];
        }
    }
    __syncthreads();

    // LayerNorm: each wave normalizes 2 of the 16 rows
#pragma unroll
    for (int rr = 0; rr < 2; ++rr) {
        int m = wave * 2 + rr;
        float s = 0.0f, s2 = 0.0f;
        for (int c = lane; c < DIM; c += 32) {
            float x = tile[m][c];
            s += x; s2 += x * x;
        }
#pragma unroll
        for (int off = 16; off; off >>= 1) {
            s  += __shfl_xor(s,  off, 32);
            s2 += __shfl_xor(s2, off, 32);
        }
        float mu  = s * (1.0f / DIM);
        float var = s2 * (1.0f / DIM) - mu * mu;
        float inv = rsqrtf(var + 1e-5f);
        for (int c = lane; c < DIM; c += 32) {
            float x = tile[m][c];
            out[(size_t)(row0 + m) * DIM + c] = (x - mu) * inv * gamma[c] + beta[c];
        }
    }
}

// ---------- host ----------
extern "C" void kernel_launch(void* const* d_in, const int* in_sizes, int n_in,
                              void* d_out, int out_size, void* d_ws, size_t ws_size,
                              hipStream_t stream) {
    const float* h_src   = (const float*)d_in[0];
    const float* h_dst   = (const float*)d_in[1];
    const int*   src_idx = (const int*)d_in[2];
    const int*   dst_idx = (const int*)d_in[3];
    const float* Wq = (const float*)d_in[4];
    const float* bq = (const float*)d_in[5];
    const float* Wk = (const float*)d_in[6];
    const float* bk = (const float*)d_in[7];
    const float* Wv = (const float*)d_in[8];
    const float* bv = (const float*)d_in[9];
    const float* Wp = (const float*)d_in[10];
    const float* bp = (const float*)d_in[11];
    const float* gamma = (const float*)d_in[12];
    const float* beta  = (const float*)d_in[13];
    float* out = (float*)d_out;

    const int N  = in_sizes[0] / DIM;
    const int E  = in_sizes[2];
    const int ND = N * DIM;
    const int WD = DIM * DIM;

    char* ws = (char*)d_ws;
    size_t off = 0;
    auto carve = [&](size_t bytes) -> void* {
        void* p = ws + off;
        off = (off + bytes + 255) & ~(size_t)255;
        return p;
    };
    unsigned short* hsrc_bf = (unsigned short*)carve((size_t)ND * 2);
    unsigned short* hdst_bf = (unsigned short*)carve((size_t)ND * 2);
    unsigned short* wq_bf   = (unsigned short*)carve((size_t)WD * 2);
    unsigned short* wk_bf   = (unsigned short*)carve((size_t)WD * 2);
    unsigned short* wv_bf   = (unsigned short*)carve((size_t)WD * 2);
    unsigned short* wp_bf   = (unsigned short*)carve((size_t)WD * 2);
    unsigned short* q_bf    = (unsigned short*)carve((size_t)ND * 2);
    unsigned short* k_bf    = (unsigned short*)carve((size_t)ND * 2);
    unsigned short* v_bf    = (unsigned short*)carve((size_t)ND * 2);
    unsigned short* agg_bf  = (unsigned short*)carve((size_t)ND * 2);
    float* score  = (float*)carve((size_t)E * HEADS * 4);
    int*   cnt    = (int*)carve((size_t)N * 4);
    int*   rowptr = (int*)carve((size_t)(N + 1) * 4);
    int*   eidx   = (int*)carve((size_t)E * 4);

    // 1) bf16 conversions
    cvt_f32_bf16<<<(ND + 255) / 256, 256, 0, stream>>>(h_src, hsrc_bf, ND);
    cvt_f32_bf16<<<(ND + 255) / 256, 256, 0, stream>>>(h_dst, hdst_bf, ND);
    cvt_f32_bf16<<<(WD + 255) / 256, 256, 0, stream>>>(Wq, wq_bf, WD);
    cvt_f32_bf16<<<(WD + 255) / 256, 256, 0, stream>>>(Wk, wk_bf, WD);
    cvt_f32_bf16<<<(WD + 255) / 256, 256, 0, stream>>>(Wv, wv_bf, WD);
    cvt_f32_bf16<<<(WD + 255) / 256, 256, 0, stream>>>(Wp, wp_bf, WD);

    // 2) CSR by destination: zero -> histogram -> scan (re-zeroes cnt) -> scatter
    zero_i32<<<(N + 255) / 256, 256, 0, stream>>>(cnt, N);
    hist_dst<<<(E + 255) / 256, 256, 0, stream>>>(dst_idx, cnt, E);
    scan_excl<<<1, 256, 0, stream>>>(cnt, rowptr, N);
    fill_csr<<<(E + 255) / 256, 256, 0, stream>>>(dst_idx, rowptr, cnt, eidx, E);

    // 3) fused q/k/v projections (WMMA bf16, LDS-staged A tiles)
    dim3 ggrid(N / 16, 1, 3);
    qkv_gemm<<<ggrid, 256, 0, stream>>>(hsrc_bf, hdst_bf, wq_bf, wk_bf, wv_bf,
                                        bq, bk, bv, q_bf, k_bf, v_bf);

    // 4) edge scores (L2-resident bf16 gathers)
    edge_score<<<(E + 7) / 8, 256, 0, stream>>>(k_bf, q_bf, src_idx, dst_idx, score, E);

    // 5) per-destination softmax + aggregation, atomic-free, emits bf16 rows
    dst_agg<<<(N + 7) / 8, 256, 0, stream>>>(v_bf, src_idx, eidx, rowptr, score, agg_bf, N);

    // 6) output projection + residual + LayerNorm (WMMA, fused)
    proj_ln<<<N / 16, 256, 0, stream>>>(agg_bf, wp_bf, bp, h_dst, gamma, beta, out);
}
